// IntraGraphAttention_3478923509961
// MI455X (gfx1250) — matrix operand order repaired and checked by reading, not compiled, not run on hardware
//
#include <hip/hip_runtime.h>
#include <hip/hip_bf16.h>
#include <math.h>

// ---------------------------------------------------------------------------
// GAT layer (IntraGraphAttention) for MI455X / gfx1250.
//   IN_DIM=128, HEADS=2, OUT_CH=32  -> projected dim HC = 64
// Pipeline:
//   h = elu(x) @ W            (WMMA f32<-f16, 16x16x32 tiles, K unrolled x4)
//   a_src/a_dst per node      (small dots)
//   edge pass 1: segment max of leaky-relu logits (float atomic max)
//   edge pass 2: unnormalized softmax aggregate (wave-per-edge, f32 atomics)
//   finalize: out /= denom, += bias
// ---------------------------------------------------------------------------

typedef __attribute__((ext_vector_type(16))) _Float16 v16h;
typedef __attribute__((ext_vector_type(8)))  float    v8f;

#define IN_DIM 128
#define HC     64
#define NEG_SLOPE 0.2f

// ---------------- init: out=0, denom=0, segmax=-inf ------------------------
__global__ void gat_init(float* __restrict__ out, float* __restrict__ segmax,
                         float* __restrict__ denom, int nout, int nseg) {
    int t = blockIdx.x * blockDim.x + threadIdx.x;
    if (t < nout) out[t] = 0.0f;
    if (t < nseg) { segmax[t] = -1e30f; denom[t] = 0.0f; }
}

// ---------------- WMMA GEMM: h = elu(x) @ W --------------------------------
// One wave computes one 16(row) x 16(col) tile of h, K unrolled over 4 steps
// of 32. Block = 128 threads = 4 waves -> the 4 column tiles of HC=64.
// Tail row-tile is clamped so every tile is fully in range: overlapping rows
// are recomputed with identical values (deterministic), no epilogue branches,
// EXEC stays all-ones throughout (WMMA requirement).
__global__ __launch_bounds__(128) void gat_gemm_wmma(
    const float* __restrict__ x, const float* __restrict__ W,
    float* __restrict__ h, int N) {
    const int lane = threadIdx.x & 31;
    const int wave = threadIdx.x >> 5;          // 0..3 column tile
    int row0 = blockIdx.x * 16;
    if (row0 + 16 > N) row0 = N - 16;           // clamp tail tile (N >= 16)
    const int col0 = wave * 16;

    const int m    = lane & 15;                 // row within tile / N within tile
    const int kgrp = (lane >> 4) & 1;           // lane-half selects K subgroup

    v8f c = {};
    #pragma unroll
    for (int kb = 0; kb < IN_DIM; kb += 32) {
        // ---- A tile 16x32 f16: lanes 0-15 hold K=0..15 (half j <-> K=j),
        //      lanes 16-31 hold K=16..31 (i.e. +16 offset).
        v16h a;
        const float* xr = x + (size_t)(row0 + m) * IN_DIM + kb + kgrp * 16;
        #pragma unroll
        for (int j = 0; j < 16; ++j) {
            int vg = j >> 1, pos = j & 1;
            int k = vg * 2 + pos;
            float v = xr[k];
            v = (v > 0.0f) ? v : (__expf(v) - 1.0f);   // fused ELU
            a[j] = (_Float16)v;
        }
        // ---- B tile 32x16 f16: lanes 0-15 hold K=0..15 (half j <-> K=j),
        //      lanes 16-31 hold K=16..31; N = lane&15.
        v16h b;
        const float* wr = W + (size_t)(kb + kgrp * 16) * HC + col0 + m;
        #pragma unroll
        for (int j = 0; j < 16; ++j) {
            b[j] = (_Float16)wr[(size_t)j * HC];
        }
        c = __builtin_amdgcn_wmma_f32_16x16x32_f16(
                /*neg_a=*/false, a, /*neg_b=*/false, b,
                /*c_mod=*/(short)0, c, /*reuse_a=*/false, /*reuse_b=*/false);
    }
    // ---- C tile 16x16 f32: lane<16 -> (M=v, N=lane); lane>=16 -> (M=v+8).
    float* hp = h + (size_t)(row0 + kgrp * 8) * HC + col0 + m;
    #pragma unroll
    for (int v = 0; v < 8; ++v) {
        hp[(size_t)v * HC] = c[v];
    }
}

// ---------------- per-node attention halves --------------------------------
__global__ void gat_att(const float* __restrict__ h,
                        const float* __restrict__ att_src,
                        const float* __restrict__ att_dst,
                        float* __restrict__ a_src, float* __restrict__ a_dst,
                        int N) {
    int t = blockIdx.x * blockDim.x + threadIdx.x;    // t = n*2 + head
    if (t >= N * 2) return;
    int n = t >> 1, hd = t & 1;
    const float* hp = h + (size_t)n * HC + hd * 32;
    const float* as = att_src + hd * 32;
    const float* ad = att_dst + hd * 32;
    float s1 = 0.0f, s2 = 0.0f;
    #pragma unroll
    for (int c = 0; c < 32; ++c) {
        float v = hp[c];
        s1 += v * as[c];
        s2 += v * ad[c];
    }
    a_src[t] = s1;
    a_dst[t] = s2;
}

// float atomic max via int punning (works for mixed signs; init = -1e30)
__device__ __forceinline__ void atomicMaxFloat(float* addr, float val) {
    if (val >= 0.0f) atomicMax((int*)addr, __float_as_int(val));
    else             atomicMin((unsigned int*)addr, (unsigned int)__float_as_int(val));
}

// ---------------- edge pass 1: segment max of logits -----------------------
__global__ void gat_edge_max(const int* __restrict__ src, const int* __restrict__ dst,
                             const float* __restrict__ a_src, const float* __restrict__ a_dst,
                             float* __restrict__ segmax, int E, int N) {
    int e = blockIdx.x * blockDim.x + threadIdx.x;
    if (e >= E + N) return;
    int s, d;
    if (e < E) { s = src[e]; d = dst[e]; } else { s = d = e - E; }  // self loops
    #pragma unroll
    for (int hd = 0; hd < 2; ++hd) {
        float l = a_src[s * 2 + hd] + a_dst[d * 2 + hd];
        l = (l >= 0.0f) ? l : NEG_SLOPE * l;
        atomicMaxFloat(&segmax[d * 2 + hd], l);
    }
}

// ---------------- edge pass 2: weighted scatter aggregate ------------------
// One wave (32 lanes) per edge; lane = channel, handles both heads.
// src/dst loads are wave-uniform -> coalesce to a single L2 request.
__global__ __launch_bounds__(256) void gat_edge_agg(
    const int* __restrict__ src, const int* __restrict__ dst,
    const float* __restrict__ a_src, const float* __restrict__ a_dst,
    const float* __restrict__ segmax, const float* __restrict__ h,
    float* __restrict__ denom, float* __restrict__ out, int E, int N) {
    int wid  = blockIdx.x * 8 + (threadIdx.x >> 5);   // edge id (8 waves/block)
    int lane = threadIdx.x & 31;
    if (wid >= E + N) return;
    int s, d;
    if (wid < E) { s = src[wid]; d = dst[wid]; } else { s = d = wid - E; }

    float l0 = a_src[s * 2 + 0] + a_dst[d * 2 + 0];
    float l1 = a_src[s * 2 + 1] + a_dst[d * 2 + 1];
    l0 = (l0 >= 0.0f) ? l0 : NEG_SLOPE * l0;
    l1 = (l1 >= 0.0f) ? l1 : NEG_SLOPE * l1;
    float w0 = __expf(l0 - segmax[d * 2 + 0]);
    float w1 = __expf(l1 - segmax[d * 2 + 1]);

    if (lane == 0) atomicAdd(&denom[d * 2 + 0], w0);
    if (lane == 1) atomicAdd(&denom[d * 2 + 1], w1);

    // head 0: channels 0..31 ; head 1: channels 32..63  (h is L2-resident)
    atomicAdd(&out[(size_t)d * HC + lane],      w0 * h[(size_t)s * HC + lane]);
    atomicAdd(&out[(size_t)d * HC + 32 + lane], w1 * h[(size_t)s * HC + 32 + lane]);
}

// ---------------- finalize: normalize + bias -------------------------------
__global__ void gat_final(float* __restrict__ out, const float* __restrict__ denom,
                          const float* __restrict__ bias, int N) {
    int t = blockIdx.x * blockDim.x + threadIdx.x;
    if (t >= N * HC) return;
    int n = t >> 6, hc = t & 63;
    out[t] = out[t] / (denom[n * 2 + (hc >> 5)] + 1e-16f) + bias[hc];
}

// ---------------------------------------------------------------------------
extern "C" void kernel_launch(void* const* d_in, const int* in_sizes, int n_in,
                              void* d_out, int out_size, void* d_ws, size_t ws_size,
                              hipStream_t stream) {
    const float* x       = (const float*)d_in[0];
    const float* W       = (const float*)d_in[1];
    const float* att_src = (const float*)d_in[2];
    const float* att_dst = (const float*)d_in[3];
    const float* bias    = (const float*)d_in[4];
    const int*   eidx    = (const int*)d_in[5];

    const int N = in_sizes[0] / IN_DIM;     // 50000
    const int E = in_sizes[5] / 2;          // 1600000
    const int* src = eidx;
    const int* dst = eidx + E;
    float* out = (float*)d_out;

    // workspace layout (floats): h[N*64] | a_src[N*2] | a_dst[N*2] | segmax[N*2] | denom[N*2]
    float* h      = (float*)d_ws;
    float* a_src  = h      + (size_t)N * HC;
    float* a_dst  = a_src  + (size_t)N * 2;
    float* segmax = a_dst  + (size_t)N * 2;
    float* denom  = segmax + (size_t)N * 2;

    const int Etot = E + N;
    const int nout = N * HC;

    gat_init<<<(nout + 255) / 256, 256, 0, stream>>>(out, segmax, denom, nout, N * 2);
    gat_gemm_wmma<<<(N + 15) / 16, 128, 0, stream>>>(x, W, h, N);
    gat_att<<<(N * 2 + 255) / 256, 256, 0, stream>>>(h, att_src, att_dst, a_src, a_dst, N);
    gat_edge_max<<<(Etot + 255) / 256, 256, 0, stream>>>(src, dst, a_src, a_dst, segmax, E, N);
    gat_edge_agg<<<(Etot + 7) / 8, 256, 0, stream>>>(src, dst, a_src, a_dst, segmax, h, denom, out, E, N);
    gat_final<<<(nout + 255) / 256, 256, 0, stream>>>(out, denom, bias, N);
}
